// EedgePath_MPNN_44770739093675
// MI455X (gfx1250) — matrix operand order repaired and checked by reading, not compiled, not run on hardware
//
#include <hip/hip_runtime.h>
#include <hip/hip_bf16.h>
#include <math.h>
#include <stdint.h>

// Problem constants (match the reference)
#define NNODES 25000
#define NPAD   25024            // NNODES rounded up to 64-row tiles (391 blocks)
#define DEG    8
#define NEDGES (NNODES * DEG)   // 200000 (multiple of 32 and 64 -> no edge guards)
#define H      64
#define NLAYERS 3
#define EPSF   1e-8f

typedef __attribute__((ext_vector_type(16))) _Float16 v16h;
typedef __attribute__((ext_vector_type(8)))  float    v8f;

// ---------------------------------------------------------------------------
// WMMA fragment helpers (CDNA5 v_wmma_f32_16x16x32_f16, wave32).
// A (16x32 f16): lane m=lane&15, lanes 0-15 hold K 0..7 & 16..23, lanes 16-31
// hold K 8..15 & 24..31 (ISA 7.12.2).  B (32x16): N = lane&15, 16 halves = K in A's order.
// C/D (16x16 f32): VGPR r -> M = r + 8*(lane>>4), N = lane&15.
// ---------------------------------------------------------------------------
__device__ __forceinline__ v16h load_a_frag(const _Float16* sA, int ld, int k0, int lane) {
  const int m = lane & 15;
  const int half = lane >> 4;
  const _Float16* row = sA + m * ld + k0 + half * 8;
  v16h a;
#pragma unroll
  for (int j = 0; j < 16; ++j) {
    const int v = j >> 1;
    const int kk = ((v < 4) ? (2 * v) : (16 + 2 * (v - 4))) + (j & 1);
    a[j] = row[kk];
  }
  return a;
}

__device__ __forceinline__ v16h load_b_frag(const _Float16* sW, int ldn, int k0, int n0, int lane) {
  const _Float16* col = sW + (size_t)(k0 + (lane >> 4) * 8) * ldn + n0 + (lane & 15);
  v16h b;
#pragma unroll
  for (int j = 0; j < 16; ++j) b[j] = col[(size_t)((j < 8) ? j : (16 + (j - 8))) * ldn];
  return b;
}

__device__ __forceinline__ v8f wmma16(v16h a, v16h b, v8f c) {
  return __builtin_amdgcn_wmma_f32_16x16x32_f16(false, a, false, b, (short)0, c, false, false);
}

// CDNA5 async copy: 16 bytes per lane, global (SGPR base + VGPR byte offset,
// GVS mode) -> LDS[vdst]. Tracked by ASYNCcnt; consumer does s_wait_asynccnt 0
// before the workgroup barrier.
__device__ __forceinline__ void async_copy_b128(uint32_t lds_addr, const void* gbase,
                                                uint32_t byte_off) {
  asm volatile("global_load_async_to_lds_b128 %0, %1, %2 offset:0"
               :: "v"(lds_addr), "v"(byte_off), "s"(gbase)
               : "memory");
}
__device__ __forceinline__ void async_wait0() {
  asm volatile("s_wait_asynccnt 0" ::: "memory");
}
__device__ __forceinline__ uint32_t lds_addr32(const void* p) {
  return (uint32_t)(uintptr_t)p;   // low 32 bits of flat shared ptr == LDS offset
}

// ---------------------------------------------------------------------------
// Utility kernels
// ---------------------------------------------------------------------------
__global__ void k_zero(float* __restrict__ p, int n) {
  int i = blockIdx.x * blockDim.x + threadIdx.x;
  if (i < n) p[i] = 0.f;
}

// Pack fp32 weights -> f16, concatenated along K for fused GEMMs.
__global__ void k_pack_node(const float* __restrict__ Ws, const float* __restrict__ Wm,
                            const float* __restrict__ We, _Float16* __restrict__ o) {
  int i = blockIdx.x * blockDim.x + threadIdx.x;
  if (i >= NLAYERS * H * H) return;
  int l = i / (H * H), rc = i - l * (H * H);
  int r = rc >> 6, c = rc & 63;
  _Float16* dst = o + (size_t)l * (3 * H * H);
  dst[(r)*H + c]           = (_Float16)Ws[i];
  dst[(H + r) * H + c]     = (_Float16)Wm[i];
  dst[(2 * H + r) * H + c] = (_Float16)We[i];
}

__global__ void k_pack_fc(const float* __restrict__ Wfc, _Float16* __restrict__ o) {
  int i = blockIdx.x * blockDim.x + threadIdx.x;
  if (i >= NLAYERS * 256 * H) return;
  int l = i / (256 * H), rc = i - l * (256 * H);
  o[i] = (_Float16)Wfc[(size_t)l * 257 * H + rc];  // rows 0..255 (sim row 256 stays fp32)
}

__global__ void k_pack_edge(const float* __restrict__ Ws, const float* __restrict__ Wn,
                            _Float16* __restrict__ o) {
  int i = blockIdx.x * blockDim.x + threadIdx.x;
  if (i >= NLAYERS * H * H) return;
  int l = i / (H * H), rc = i - l * (H * H);
  int r = rc >> 6, c = rc & 63;
  _Float16* dst = o + (size_t)l * (2 * H * H);
  dst[r * H + c]       = (_Float16)Ws[i];
  dst[(H + r) * H + c] = (_Float16)Wn[i];
}

// ---------------------------------------------------------------------------
// Segment sums by edge_trg (fp32 L2 atomics). edge_src[e] == e>>3 by design.
// Grid exactly covers NEDGES*32 lanes -> no guards.
// ---------------------------------------------------------------------------
__global__ void k_scatter_node(const float* __restrict__ x, const float* __restrict__ ef,
                               const int* __restrict__ trg,
                               float* __restrict__ agg_x, float* __restrict__ agg_e) {
  long gid = (long)blockIdx.x * blockDim.x + threadIdx.x;
  int e = (int)(gid >> 5);
  int c = ((int)gid & 31) * 2;
  int t = trg[e];
  const float2 xv = *(const float2*)(x + (size_t)(e >> 3) * H + c);
  const float2 ev = *(const float2*)(ef + (size_t)e * H + c);
  float* ax = agg_x + (size_t)t * H + c;
  float* ae = agg_e + (size_t)t * H + c;
  atomicAdd(ax + 0, xv.x); atomicAdd(ax + 1, xv.y);
  atomicAdd(ae + 0, ev.x); atomicAdd(ae + 1, ev.y);
}

__global__ void k_scatter_line(const float* __restrict__ ef, const int* __restrict__ trg,
                               float* __restrict__ agg) {
  long gid = (long)blockIdx.x * blockDim.x + threadIdx.x;
  int e = (int)(gid >> 5);
  int c = ((int)gid & 31) * 2;
  int t = trg[e];
  const float2 ev = *(const float2*)(ef + (size_t)e * H + c);
  float* ag = agg + (size_t)t * H + c;
  atomicAdd(ag + 0, ev.x); atomicAdd(ag + 1, ev.y);
}

// ---------------------------------------------------------------------------
// Node update: xcur = [x | agg_x | agg_e] (64x192) @ Wn (192x64) + bn
// 8 waves, 16 tiles of 16x16 -> 2 tiles/wave, K=192 -> 6 WMMA per tile.
// Output buffer is padded to NPAD rows -> unguarded stores.
// ---------------------------------------------------------------------------
__global__ void __launch_bounds__(256)
k_node_gemm(const float* __restrict__ x, const float* __restrict__ ax,
            const float* __restrict__ ae, const _Float16* __restrict__ Wn,
            const float* __restrict__ bn, float* __restrict__ out) {
  __shared__ _Float16 sA[64 * 192];   // 24 KB
  __shared__ _Float16 sW[192 * 64];   // 24 KB
  const int tid = threadIdx.x;
  const int row0 = blockIdx.x * 64;

  // Async DMA the f16 weight tile into LDS (24 KB = 1536 x b128, 6 per thread).
  {
    const uint32_t lds0 = lds_addr32(&sW[0]);
    for (int i = tid; i < (192 * 64 * 2) / 16; i += 256)
      async_copy_b128(lds0 + i * 16, Wn, (uint32_t)i * 16);
  }
  for (int i = tid; i < 64 * 192; i += 256) {
    int r = i / 192, c = i - r * 192;
    int row = row0 + r;
    float v = 0.f;
    if (row < NNODES) {
      v = (c < 64)  ? x [(size_t)row * H + c]
        : (c < 128) ? ax[(size_t)row * H + (c - 64)]
                    : ae[(size_t)row * H + (c - 128)];
    }
    sA[i] = (_Float16)v;
  }
  async_wait0();
  __syncthreads();

  const int lane = tid & 31, wave = tid >> 5, half = lane >> 4;
  for (int tIdx = wave; tIdx < 16; tIdx += 8) {
    const int rt = tIdx >> 2, ct = tIdx & 3;
    v8f acc = {};
#pragma unroll
    for (int k0 = 0; k0 < 192; k0 += 32) {
      v16h a = load_a_frag(sA + rt * 16 * 192, 192, k0, lane);
      v16h b = load_b_frag(sW, 64, k0, ct * 16, lane);
      acc = wmma16(a, b, acc);
    }
    const int n = ct * 16 + (lane & 15);
    const float bias = bn[n];
    float* orow = out + (size_t)(row0 + rt * 16 + half * 8) * H + n;
#pragma unroll
    for (int r = 0; r < 8; ++r) orow[(size_t)r * H] = acc[r] + bias;
  }
}

// Row L2 norms (pre-relu, for cosine sim) + relu'd copy for the next layer.
__global__ void k_norm_relu(const float* __restrict__ xc, float* __restrict__ xr,
                            float* __restrict__ norms) {
  int gid = blockIdx.x * blockDim.x + threadIdx.x;
  int row = gid >> 4;
  if (row >= NNODES) return;
  int c = (gid & 15) * 4;
  const float4 v = *(const float4*)(xc + (size_t)row * H + c);
  float s = v.x * v.x + v.y * v.y + v.z * v.z + v.w * v.w;
  s += __shfl_xor(s, 1, 32); s += __shfl_xor(s, 2, 32);
  s += __shfl_xor(s, 4, 32); s += __shfl_xor(s, 8, 32);
  if ((gid & 15) == 0) norms[row] = sqrtf(s);
  float4 r = make_float4(fmaxf(v.x, 0.f), fmaxf(v.y, 0.f), fmaxf(v.z, 0.f), fmaxf(v.w, 0.f));
  *(float4*)(xr + (size_t)row * H + c) = r;
}

// ---------------------------------------------------------------------------
// Edge FC: eout = [ef | xs | xt | |xs-xt|] (32x256) @ Wfc16 + sim*Wfc[256] + bfc
// 32 edges/block, 8 waves, 8 tiles -> 1 tile/wave, K=256 -> 8 WMMA per tile.
// NEDGES % 32 == 0 -> no guards. In-place safe (rows staged in LDS first).
// ---------------------------------------------------------------------------
__global__ void __launch_bounds__(256)
k_edge_fc(const float* __restrict__ ein, const float* __restrict__ xc,
          const int* __restrict__ trg, const float* __restrict__ norms,
          const _Float16* __restrict__ Wfc16, const float* __restrict__ WfcSim,
          const float* __restrict__ bfc, float* __restrict__ eout) {
  __shared__ _Float16 sA[32 * 256];   // 16 KB
  __shared__ _Float16 sW[256 * 64];   // 32 KB
  __shared__ float sSim[32];
  const int tid = threadIdx.x;
  const int e0 = blockIdx.x * 32;

  // Async DMA the f16 FC weights (32 KB = 2048 x b128, 8 per thread).
  {
    const uint32_t lds0 = lds_addr32(&sW[0]);
    for (int i = tid; i < (256 * 64 * 2) / 16; i += 256)
      async_copy_b128(lds0 + i * 16, Wfc16, (uint32_t)i * 16);
  }

  // Build A: one 16-lane group per edge row (2 rows per group).
  for (int rr = tid >> 4; rr < 32; rr += 16) {
    const int e = e0 + rr;
    const int c = (tid & 15) * 4;
    const int s = e >> 3;          // edge_src[e]
    const int t = trg[e];
    const float4 fe = *(const float4*)(ein + (size_t)e * H + c);
    const float4 xs = *(const float4*)(xc + (size_t)s * H + c);
    const float4 xt = *(const float4*)(xc + (size_t)t * H + c);
    _Float16* rowA = sA + rr * 256;
    rowA[c + 0] = (_Float16)fe.x; rowA[c + 1] = (_Float16)fe.y;
    rowA[c + 2] = (_Float16)fe.z; rowA[c + 3] = (_Float16)fe.w;
    rowA[64 + c + 0] = (_Float16)xs.x; rowA[64 + c + 1] = (_Float16)xs.y;
    rowA[64 + c + 2] = (_Float16)xs.z; rowA[64 + c + 3] = (_Float16)xs.w;
    rowA[128 + c + 0] = (_Float16)xt.x; rowA[128 + c + 1] = (_Float16)xt.y;
    rowA[128 + c + 2] = (_Float16)xt.z; rowA[128 + c + 3] = (_Float16)xt.w;
    rowA[192 + c + 0] = (_Float16)fabsf(xs.x - xt.x);
    rowA[192 + c + 1] = (_Float16)fabsf(xs.y - xt.y);
    rowA[192 + c + 2] = (_Float16)fabsf(xs.z - xt.z);
    rowA[192 + c + 3] = (_Float16)fabsf(xs.w - xt.w);
    float d = xs.x * xt.x + xs.y * xt.y + xs.z * xt.z + xs.w * xt.w;
    d += __shfl_xor(d, 1, 32); d += __shfl_xor(d, 2, 32);
    d += __shfl_xor(d, 4, 32); d += __shfl_xor(d, 8, 32);
    if ((tid & 15) == 0) sSim[rr] = d / fmaxf(norms[s] * norms[t], EPSF);
  }
  async_wait0();
  __syncthreads();

  const int lane = tid & 31, wave = tid >> 5, half = lane >> 4;
  const int rt = wave >> 2, ct = wave & 3;   // 2 x 4 tiles, one per wave
  v8f acc = {};
#pragma unroll
  for (int k0 = 0; k0 < 256; k0 += 32) {
    v16h a = load_a_frag(sA + rt * 16 * 256, 256, k0, lane);
    v16h b = load_b_frag(sW, 64, k0, ct * 16, lane);
    acc = wmma16(a, b, acc);
  }
  const int n = ct * 16 + (lane & 15);
  const float bias = bfc[n];
  const float wsim = WfcSim[n];   // fp32 row 256 of Wfc (sim column)
  const int m0 = rt * 16 + half * 8;
  float* orow = eout + (size_t)(e0 + m0) * H + n;
#pragma unroll
  for (int r = 0; r < 8; ++r) orow[(size_t)r * H] = acc[r] + sSim[m0 + r] * wsim + bias;
}

// ---------------------------------------------------------------------------
// Edge conv: eout = [ef | node_agg[e>>3]] (64x128) @ We (128x64) + be
// agg_le[e] == node_agg[src[e]] == node_agg[e>>3] (line-graph collapse).
// NEDGES % 64 == 0 -> no guards.
// ---------------------------------------------------------------------------
__global__ void __launch_bounds__(256)
k_edge_gemm(const float* __restrict__ ef, const float* __restrict__ agg,
            const _Float16* __restrict__ We, const float* __restrict__ be,
            float* __restrict__ out) {
  __shared__ _Float16 sA[64 * 128];   // 16 KB
  __shared__ _Float16 sW[128 * 64];   // 16 KB
  const int tid = threadIdx.x;
  const int e0 = blockIdx.x * 64;

  // Async DMA the f16 edge weights (16 KB = 1024 x b128, 4 per thread).
  {
    const uint32_t lds0 = lds_addr32(&sW[0]);
    for (int i = tid; i < (128 * 64 * 2) / 16; i += 256)
      async_copy_b128(lds0 + i * 16, We, (uint32_t)i * 16);
  }
  for (int i = tid; i < 64 * 128; i += 256) {
    int r = i >> 7, c = i & 127;
    int e = e0 + r;
    float v = (c < 64) ? ef[(size_t)e * H + c] : agg[(size_t)(e >> 3) * H + (c - 64)];
    sA[i] = (_Float16)v;
  }
  async_wait0();
  __syncthreads();

  const int lane = tid & 31, wave = tid >> 5, half = lane >> 4;
  for (int tIdx = wave; tIdx < 16; tIdx += 8) {
    const int rt = tIdx >> 2, ct = tIdx & 3;
    v8f acc = {};
#pragma unroll
    for (int k0 = 0; k0 < 128; k0 += 32) {
      v16h a = load_a_frag(sA + rt * 16 * 128, 128, k0, lane);
      v16h b = load_b_frag(sW, 64, k0, ct * 16, lane);
      acc = wmma16(a, b, acc);
    }
    const int n = ct * 16 + (lane & 15);
    const float bias = be[n];
    float* orow = out + (size_t)(e0 + rt * 16 + half * 8) * H + n;
#pragma unroll
    for (int r = 0; r < 8; ++r) orow[(size_t)r * H] = acc[r] + bias;
  }
}

// ---------------------------------------------------------------------------
extern "C" void kernel_launch(void* const* d_in, const int* in_sizes, int n_in,
                              void* d_out, int out_size, void* d_ws, size_t ws_size,
                              hipStream_t stream) {
  (void)in_sizes; (void)n_in; (void)out_size; (void)ws_size;
  const float* d_x   = (const float*)d_in[0];
  const float* d_ef  = (const float*)d_in[1];
  const float* d_WnS = (const float*)d_in[2];
  const float* d_WnM = (const float*)d_in[3];
  const float* d_WnE = (const float*)d_in[4];
  const float* d_bn  = (const float*)d_in[5];
  const float* d_Wfc = (const float*)d_in[6];
  const float* d_bfc = (const float*)d_in[7];
  const float* d_WeS = (const float*)d_in[8];
  const float* d_WeN = (const float*)d_in[9];
  const float* d_be  = (const float*)d_in[10];
  const int*   d_trg = (const int*)d_in[12];
  // d_in[11] (edge_src == e>>3), d_in[13]/[14] (line graph) are implicit.
  float* out = (float*)d_out;

  char* ws = (char*)d_ws;
  auto carve = [&](size_t bytes) -> char* {
    char* p = ws; ws += (bytes + 255) & ~(size_t)255; return p;
  };
  float* ebuf  = (float*)carve((size_t)NEDGES * H * 4);
  float* xcur  = (float*)carve((size_t)NPAD * H * 4);    // padded: unguarded stores
  float* xrel  = (float*)carve((size_t)NPAD * H * 4);
  float* aggx  = (float*)carve((size_t)NNODES * H * 4);  // also line-graph node agg
  float* agge  = (float*)carve((size_t)NNODES * H * 4);
  float* norms = (float*)carve((size_t)NNODES * 4);
  _Float16* Wn16 = (_Float16*)carve((size_t)NLAYERS * 3 * H * H * 2);
  _Float16* Wf16 = (_Float16*)carve((size_t)NLAYERS * 256 * H * 2);
  _Float16* We16 = (_Float16*)carve((size_t)NLAYERS * 2 * H * H * 2);

  {
    int n = NLAYERS * H * H;
    k_pack_node<<<(n + 255) / 256, 256, 0, stream>>>(d_WnS, d_WnM, d_WnE, Wn16);
    k_pack_edge<<<(n + 255) / 256, 256, 0, stream>>>(d_WeS, d_WeN, We16);
    int m = NLAYERS * 256 * H;
    k_pack_fc<<<(m + 255) / 256, 256, 0, stream>>>(d_Wfc, Wf16);
  }

  const int NH = NNODES * H;
  const int nodeBlocks   = NPAD / 64;          // 391
  const int edgeBlocks64 = NEDGES / 64;        // 3125
  const int edgeBlocks32 = NEDGES / 32;        // 6250
  const int scatterBlocks = (NEDGES * 32) / 256;  // exact

  for (int l = 0; l < NLAYERS; ++l) {
    const float* xin = (l == 0) ? d_x  : xrel;
    const float* ein = (l == 0) ? d_ef : ebuf;
    k_zero<<<(NH + 255) / 256, 256, 0, stream>>>(aggx, NH);
    k_zero<<<(NH + 255) / 256, 256, 0, stream>>>(agge, NH);
    k_scatter_node<<<scatterBlocks, 256, 0, stream>>>(xin, ein, d_trg, aggx, agge);
    k_node_gemm<<<nodeBlocks, 256, 0, stream>>>(
        xin, aggx, agge, Wn16 + (size_t)l * 3 * H * H, d_bn + (size_t)l * H, xcur);
    k_norm_relu<<<(NNODES * 16 + 255) / 256, 256, 0, stream>>>(xcur, xrel, norms);
    k_edge_fc<<<edgeBlocks32, 256, 0, stream>>>(
        ein, xcur, d_trg, norms, Wf16 + (size_t)l * 256 * H,
        d_Wfc + ((size_t)l * 257 + 256) * H, d_bfc + (size_t)l * H, ebuf);
    k_zero<<<(NH + 255) / 256, 256, 0, stream>>>(aggx, NH);
    k_scatter_line<<<scatterBlocks, 256, 0, stream>>>(ebuf, d_trg, aggx);
    float* eo = (l == NLAYERS - 1) ? out : ebuf;
    k_edge_gemm<<<edgeBlocks64, 256, 0, stream>>>(
        ebuf, aggx, We16 + (size_t)l * 2 * H * H, d_be + (size_t)l * H, eo);
  }
}
